// EnhancedSpikingAudioNet_24060406792395
// MI455X (gfx1250) — matrix operand order repaired and checked by reading, not compiled, not run on hardware
//
#include <hip/hip_runtime.h>

// ---------------------------------------------------------------------------
// Persistent 4-layer LIF SNN for MI455X (gfx1250, wave32, WMMA).
//
//   grid  = 16 workgroups, one per 16-row batch tile (B=256)
//   block = 512 threads = 16 waves
//   Per-wave column tiles: L1: 3, L2: 2, L3: 1, L4: wave 0 only.
//   Membranes persist in v8f WMMA accumulators across all T=100 steps.
//   xt staged f32->f16 into LDS once per step (shared by all 16 waves);
//   spikes exchanged via LDS; weights pre-converted to f16 in d_ws and
//   streamed from L2 every step (an asm memory clobber stops LICM from
//   hoisting 160 weight fragments into scratch spills).
//   Dynamic LDS (~83 KB) -> hipFuncSetAttribute before launch.
// ---------------------------------------------------------------------------

typedef _Float16 half_t;
typedef __attribute__((ext_vector_type(16))) _Float16 v16h;
typedef __attribute__((ext_vector_type(8)))  _Float16 v8h;
typedef __attribute__((ext_vector_type(4)))  _Float16 v4h;
typedef __attribute__((ext_vector_type(8)))  float    v8f;

#define SNN_BETA 0.95f
#define SNN_THR  1.0f

#define T_STEPS 100
#define DIM_D   1024
#define DIM_H1  768
#define DIM_H2  512
#define DIM_H3  256
#define DIM_C   10

// LDS row strides (halves); +8 halves (16B) pad keeps rows on distinct bank
// groups while every fragment chunk stays 16B aligned.
#define XS  (DIM_D  + 8)
#define S1S (DIM_H1 + 8)
#define S2S (DIM_H2 + 8)
#define S3S (DIM_H3 + 8)

// LDS layout offsets in halves.
#define OFF_XH 0
#define OFF_S1 (16 * XS)
#define OFF_S2 (OFF_S1 + 16 * S1S)
#define OFF_S3 (OFF_S2 + 16 * S2S)
#define LDS_HALVES (OFF_S3 + 16 * S3S)
#define LDS_BYTES  (LDS_HALVES * 2)          // 82,944 B

// ---------------------------------------------------------------------------
// Weight conversion kernels (run every launch; d_ws is not preserved).
// ---------------------------------------------------------------------------
__global__ void snn_cvt_f16(const float* __restrict__ src,
                            half_t* __restrict__ dst, int n) {
  int i = blockIdx.x * blockDim.x + threadIdx.x;
  if (i < n) dst[i] = (half_t)src[i];
}

// W4: [10][256] f32 -> [16][256] f16, rows 10..15 zero-padded.
__global__ void snn_cvt_w4(const float* __restrict__ src,
                           half_t* __restrict__ dst) {
  int i = blockIdx.x * blockDim.x + threadIdx.x;   // 0..4095
  int r = i >> 8;
  int c = i & 255;
  dst[i] = (r < DIM_C) ? (half_t)src[r * 256 + c] : (half_t)0.0f;
}

// ---------------------------------------------------------------------------
// Fragment loaders (layouts per CDNA5 ISA 7.12.2, wave32).
// ---------------------------------------------------------------------------

// A fragment 16x32 f16 from an LDS activation buffer [16][stride] (row-major).
// lane<16: halves 0..7 = K 0..7,  halves 8..15 = K 16..23 (row = lane)
// lane>=16: halves 0..7 = K 8..15, halves 8..15 = K 24..31 (row = lane-16)
__device__ __forceinline__ v16h load_a_lds(const half_t* lds, int stride,
                                           int lane, int kTile) {
  int row = lane & 15;
  int hi  = lane >> 4;
  int base = row * stride + kTile * 32 + hi * 8;
  union { v16h v; v8h h[2]; } u;
  u.h[0] = *(const v8h*)(lds + base);
  u.h[1] = *(const v8h*)(lds + base + 16);
  return u.v;
}

// B fragment 32x16 f16 from global weights stored [N][K] row-major (== W as
// given: [fout][fin], so B[k][n] = W[n][k] needs no transpose).
// col = lane&15 ; lane<16 holds K 0..7,16..23, lane>=16 holds K 8..15,24..31.
__device__ __forceinline__ v16h load_b_w(const half_t* __restrict__ W, int K,
                                         int lane, int nTile, int kTile) {
  int col = lane & 15;
  int hi  = lane >> 4;
  const half_t* p = W + (size_t)(nTile * 16 + col) * K + kTile * 32 + hi * 8;
  union { v16h v; v8h h[2]; } u;
  u.h[0] = *(const v8h*)(p);
  u.h[1] = *(const v8h*)(p + 16);
  return u.v;
}

// NT adjacent 16x16 output tiles sharing one A-fragment load per k-step.
// Multiple independent accumulators -> ILP between B loads and the WMMA pipe.
template <int NT>
__device__ __forceinline__ void gemm_tiles(const half_t* aLds, int aStride,
                                           const half_t* __restrict__ W,
                                           int K, int lane, int nTile0,
                                           v8f* c) {
  const int nk = K / 32;
  for (int kt = 0; kt < nk; ++kt) {
    v16h a = load_a_lds(aLds, aStride, lane, kt);
#pragma unroll
    for (int j = 0; j < NT; ++j) {
      v16h b = load_b_w(W, K, lane, nTile0 + j, kt);
      c[j] = __builtin_amdgcn_wmma_f32_16x16x32_f16(false, a, false, b,
                                                    (short)0, c[j],
                                                    false, false);
    }
  }
}

// LIF update on an accumulator fragment + spike scatter to next LDS buffer.
// D layout: col = lane&15, row(i) = i + 8*(lane>=16).
__device__ __forceinline__ void lif_store(v8f cur, float bias, v8f& mem,
                                          half_t* sOut, int sStride,
                                          int lane, int nTile) {
  int col = (lane & 15) + nTile * 16;
  int hi  = lane >> 4;
#pragma unroll
  for (int i = 0; i < 8; ++i) {
    float m     = mem[i];
    float reset = (m > SNN_THR) ? SNN_THR : 0.0f;   // from PREVIOUS membrane
    m           = SNN_BETA * m + cur[i] + bias - reset;
    mem[i]      = m;
    sOut[(i + hi * 8) * sStride + col] = (m > SNN_THR) ? (half_t)1.0f
                                                       : (half_t)0.0f;
  }
}

// ---------------------------------------------------------------------------
// Main persistent kernel.
// ---------------------------------------------------------------------------
__global__ __launch_bounds__(512)
void snn_forward(const float* __restrict__ x,
                 const half_t* __restrict__ W1h, const float* __restrict__ b1,
                 const half_t* __restrict__ W2h, const float* __restrict__ b2,
                 const half_t* __restrict__ W3h, const float* __restrict__ b3,
                 const half_t* __restrict__ W4h, const float* __restrict__ b4,
                 float* __restrict__ out) {
  extern __shared__ __align__(16) half_t lds[];
  half_t* xh = lds + OFF_XH;   // [16][XS]  staged f16 input
  half_t* s1 = lds + OFF_S1;   // [16][S1S] layer-1 spikes
  half_t* s2 = lds + OFF_S2;   // [16][S2S] layer-2 spikes
  half_t* s3 = lds + OFF_S3;   // [16][S3S] layer-3 spikes

  const int tid   = threadIdx.x;
  const int wave  = tid >> 5;
  const int lane  = tid & 31;
  const int colL  = lane & 15;
  const int bTile = blockIdx.x;                    // 0..15 batch tile

  // Per-lane bias scalars for owned tiles (column-constant within a tile).
  float bias1[3], bias2[2], bias3, bias4;
#pragma unroll
  for (int j = 0; j < 3; ++j) bias1[j] = b1[(wave * 3 + j) * 16 + colL];
#pragma unroll
  for (int j = 0; j < 2; ++j) bias2[j] = b2[(wave * 2 + j) * 16 + colL];
  bias3 = b3[wave * 16 + colL];
  bias4 = (wave == 0 && colL < DIM_C) ? b4[colL] : 0.0f;

  // Persistent membrane state in accumulator registers.
  v8f m1[3]; v8f m2[2]; v8f m3[1]; v8f m4[1];
#pragma unroll
  for (int j = 0; j < 3; ++j) m1[j] = (v8f){};
#pragma unroll
  for (int j = 0; j < 2; ++j) m2[j] = (v8f){};
  m3[0] = (v8f){};
  m4[0] = (v8f){};

  const float* xBase = x + (size_t)bTile * 16 * DIM_D;

#pragma unroll 1
  for (int t = 0; t < T_STEPS; ++t) {
    // Opaque clobber: stops LICM from hoisting the (t-invariant) weight
    // fragment loads out of the loop, which previously spilled ~5 KB/lane
    // to scratch and turned the inner loop into scratch reloads.
    asm volatile("" ::: "memory");

    const float* xt = xBase + (size_t)t * 256 * DIM_D;

    // ---- stage xt f32 -> f16 into LDS (shared by all 16 waves).
    // 16x1024 = 4096 float4 groups; 8 per thread.
    for (int g = tid; g < 16 * DIM_D / 4; g += 512) {
      int row  = g >> 8;            // 256 groups per row
      int col4 = (g & 255) * 4;
      float4 v = *(const float4*)(xt + row * DIM_D + col4);
      v4h h;
      h[0] = (half_t)v.x; h[1] = (half_t)v.y;
      h[2] = (half_t)v.z; h[3] = (half_t)v.w;
      *(v4h*)(xh + row * XS + col4) = h;
    }
    // ---- prefetch next timestep's x tile (64 KB = 512 threads x 128 B).
    if (t + 1 < T_STEPS) {
      const char* nxt = (const char*)(xt + (size_t)256 * DIM_D);
      __builtin_prefetch(nxt + tid * 128, 0, 3);
    }
    __syncthreads();   // xh ready; also wave-0 rejoin point after layer 4

    // ---- layer 1: [16,1024] x [1024,768]; one A load feeds 3 WMMAs.
    {
      v8f c[3] = {(v8f){}, (v8f){}, (v8f){}};
      gemm_tiles<3>(xh, XS, W1h, DIM_D, lane, wave * 3, c);
#pragma unroll
      for (int j = 0; j < 3; ++j)
        lif_store(c[j], bias1[j], m1[j], s1, S1S, lane, wave * 3 + j);
    }
    __syncthreads();

    // ---- layer 2: [16,768] x [768,512]; one A load feeds 2 WMMAs.
    {
      v8f c[2] = {(v8f){}, (v8f){}};
      gemm_tiles<2>(s1, S1S, W2h, DIM_H1, lane, wave * 2, c);
#pragma unroll
      for (int j = 0; j < 2; ++j)
        lif_store(c[j], bias2[j], m2[j], s2, S2S, lane, wave * 2 + j);
    }
    __syncthreads();

    // ---- layer 3: [16,512] x [512,256]
    {
      v8f c[1] = {(v8f){}};
      gemm_tiles<1>(s2, S2S, W3h, DIM_H2, lane, wave, c);
      lif_store(c[0], bias3, m3[0], s3, S3S, lane, wave);
    }
    __syncthreads();   // s3 ready for wave 0

    // ---- layer 4: [16,256] x [256,16(pad of 10)], wave 0 only.
    // No trailing barrier: waves 1..15 run ahead to stage x(t+1) (writes xh
    // only, which nobody still reads); wave 0 rejoins at the staging barrier.
    if (wave == 0) {
      v8f c[1] = {(v8f){}};
      gemm_tiles<1>(s3, S3S, W4h, DIM_H3, lane, 0, c);
#pragma unroll
      for (int i = 0; i < 8; ++i) {
        float m     = m4[0][i];
        float reset = (m > SNN_THR) ? SNN_THR : 0.0f;
        m4[0][i]    = SNN_BETA * m + c[0][i] + bias4 - reset;
      }
    }
  }

  // Emit final layer-4 membrane: [256, 10] f32.
  if (wave == 0 && colL < DIM_C) {
    int hi = lane >> 4;
#pragma unroll
    for (int i = 0; i < 8; ++i) {
      int row = bTile * 16 + i + hi * 8;
      out[row * DIM_C + colL] = m4[0][i];
    }
  }
}

// ---------------------------------------------------------------------------
// Host launcher. Inputs: x, W1, b1, W2, b2, W3, b3, W4, b4 (all f32).
// Workspace: f16 copies of W1..W3 plus zero-padded W4 (~2.6 MB).
// ---------------------------------------------------------------------------
extern "C" void kernel_launch(void* const* d_in, const int* in_sizes, int n_in,
                              void* d_out, int out_size, void* d_ws,
                              size_t ws_size, hipStream_t stream) {
  (void)in_sizes; (void)n_in; (void)out_size; (void)ws_size;

  const float* x  = (const float*)d_in[0];
  const float* W1 = (const float*)d_in[1];
  const float* b1 = (const float*)d_in[2];
  const float* W2 = (const float*)d_in[3];
  const float* b2 = (const float*)d_in[4];
  const float* W3 = (const float*)d_in[5];
  const float* b3 = (const float*)d_in[6];
  const float* W4 = (const float*)d_in[7];
  const float* b4 = (const float*)d_in[8];
  float* out = (float*)d_out;

  const int nW1  = DIM_H1 * DIM_D;   // 786432
  const int nW2  = DIM_H2 * DIM_H1;  // 393216
  const int nW3  = DIM_H3 * DIM_H2;  // 131072
  const int nW4p = 16 * DIM_H3;      // 4096 (padded)

  half_t* W1h = (half_t*)d_ws;
  half_t* W2h = W1h + nW1;
  half_t* W3h = W2h + nW2;
  half_t* W4h = W3h + nW3;

  snn_cvt_f16<<<(nW1 + 255) / 256, 256, 0, stream>>>(W1, W1h, nW1);
  snn_cvt_f16<<<(nW2 + 255) / 256, 256, 0, stream>>>(W2, W2h, nW2);
  snn_cvt_f16<<<(nW3 + 255) / 256, 256, 0, stream>>>(W3, W3h, nW3);
  snn_cvt_w4<<<nW4p / 256, 256, 0, stream>>>(W4, W4h);

  // Dynamic LDS exceeds the 64 KB default; opt in (host call, capture-safe).
  hipFuncSetAttribute((const void*)snn_forward,
                      hipFuncAttributeMaxDynamicSharedMemorySize, LDS_BYTES);

  snn_forward<<<16, 512, LDS_BYTES, stream>>>(x, W1h, b1, W2h, b2, W3h, b3,
                                              W4h, b4, out);
}
